// _TimeLoopModel_67783173865878
// MI455X (gfx1250) — matrix operand order repaired and checked by reading, not compiled, not run on hardware
//
#include <hip/hip_runtime.h>
#include <cmath>

// Problem constants (match reference)
#define B_    64
#define T_    256
#define DIN   1024
#define DHID  2048
#define DOUT  1024
#define DECAYF 0.9f
#define NTOK  (B_ * T_)   // 16384 tokens

typedef __attribute__((ext_vector_type(16))) __bf16 v16bf;
typedef __attribute__((ext_vector_type(8)))  __bf16 v8bf;
typedef __attribute__((ext_vector_type(8)))  float  v8f;
typedef __attribute__((ext_vector_type(4)))  float  f4;

__device__ __forceinline__ void split1(float x, __bf16& h, __bf16& l) {
  h = (__bf16)x;              // RNE fp32 -> bf16
  l = (__bf16)(x - (float)h); // residual
}

// ---------------------------------------------------------------------------
// Pre-pass 0: split x (fp32) into bf16 hi/lo planes. 8 elements per thread.
// ---------------------------------------------------------------------------
__global__ __launch_bounds__(256) void k_split_x(const float* __restrict__ src,
                                                 __bf16* __restrict__ hi,
                                                 __bf16* __restrict__ lo,
                                                 int n8) {
  const int i = blockIdx.x * 256 + threadIdx.x;
  if (i >= n8) return;
  const size_t base = (size_t)i * 8;
  f4 a = *(const f4*)(src + base);
  f4 b = *(const f4*)(src + base + 4);
  const float f[8] = {a.x, a.y, a.z, a.w, b.x, b.y, b.z, b.w};
  v8bf h, l;
#pragma unroll
  for (int k = 0; k < 8; ++k) {
    __bf16 hh, ll;
    split1(f[k], hh, ll);
    h[k] = hh;
    l[k] = ll;
  }
  *(v8bf*)(hi + base) = h;
  *(v8bf*)(lo + base) = l;
}

// ---------------------------------------------------------------------------
// Pre-pass 1: transpose W[K][N] -> Wt[N][K] and split into bf16 hi/lo.
// LDS-tiled 64x64 so both global access phases stay coalesced. Runs once.
// ---------------------------------------------------------------------------
__global__ __launch_bounds__(256) void k_split_wt(const float* __restrict__ W,
                                                  int K, int N,
                                                  __bf16* __restrict__ tHi,
                                                  __bf16* __restrict__ tLo) {
  __shared__ float tile[64][65];
  const int n0 = blockIdx.x * 64;
  const int k0 = blockIdx.y * 64;
  const int tid = threadIdx.x;
  const int c  = tid & 63;
  const int r4 = tid >> 6;  // 0..3
#pragma unroll
  for (int p = 0; p < 16; ++p) {
    const int r = p * 4 + r4;
    tile[r][c] = W[(size_t)(k0 + r) * N + n0 + c];
  }
  __syncthreads();
#pragma unroll
  for (int p = 0; p < 16; ++p) {
    const int rr = p * 4 + r4;            // n offset within tile
    const float v = tile[c][rr];          // transposed read (conflict-free: 65 pad)
    __bf16 h, l;
    split1(v, h, l);
    const size_t o = (size_t)(n0 + rr) * K + k0 + c;
    tHi[o] = h;
    tLo[o] = l;
  }
}

// ---------------------------------------------------------------------------
// bf16x3 GEMM core, all operands pre-split bf16 in global memory.
// A frag (16x32): lane<16 -> M=lane, K 0..15; lane>=16 -> K 16..31
//   => two contiguous 16B loads.
// B frag (32x16, K-major Wt): lane<16 -> N=lane, K 0..15; lanes>=16 -> K 16..31
//   => one contiguous 32B load.
// No LDS, no barriers, no conversion VALU in the loop: loads + wmma only.
// ---------------------------------------------------------------------------
__device__ __forceinline__ v16bf load_a_frag(const __bf16* __restrict__ A, int lda,
                                             int row, int k0, int lane) {
  const int koff = (lane >> 4) << 4;
  const __bf16* p = A + (size_t)row * lda + k0 + koff;
  union { v16bf v; v8bf h[2]; } u;
  u.h[0] = *(const v8bf*)(p);        // K = k0+koff .. +7
  u.h[1] = *(const v8bf*)(p + 8);    // K = k0+koff+8 .. +15
  return u.v;
}

__device__ __forceinline__ v16bf load_b_frag(const __bf16* __restrict__ Bt, int ldb,
                                             int n, int k0, int lane) {
  const int kb = (lane >> 4) << 4;
  return *(const v16bf*)(Bt + (size_t)n * ldb + k0 + kb);  // 32B contiguous
}

__device__ __forceinline__ void gemm_tiles(const __bf16* __restrict__ Ahi,
                                           const __bf16* __restrict__ Alo,
                                           int lda, int ktot,
                                           const __bf16* __restrict__ Bthi,
                                           const __bf16* __restrict__ Btlo,
                                           int n0, v8f acc[4][2]) {
  const int tid  = threadIdx.x;
  const int lane = tid & 31;
  const int wave = tid >> 5;
  const int wm = wave >> 1;  // 0..3
  const int wn = wave & 1;   // 0..1
  const int ar = lane & 15;

#pragma unroll 1
  for (int k0 = 0; k0 < ktot; k0 += 32) {
    v16bf bh[2], bl[2];
#pragma unroll
    for (int ni = 0; ni < 2; ++ni) {
      const int n = n0 + (wn * 2 + ni) * 16 + (lane & 15);
      bh[ni] = load_b_frag(Bthi, ktot, n, k0, lane);
      bl[ni] = load_b_frag(Btlo, ktot, n, k0, lane);
    }
#pragma unroll
    for (int mi = 0; mi < 4; ++mi) {
      const int row = (wm * 4 + mi) * 16 + ar;
      const v16bf ah = load_a_frag(Ahi, lda, row, k0, lane);
      const v16bf al = load_a_frag(Alo, lda, row, k0, lane);
#pragma unroll
      for (int ni = 0; ni < 2; ++ni) {
        acc[mi][ni] = __builtin_amdgcn_wmma_f32_16x16x32_bf16(
            false, ah, false, bh[ni], (short)0, acc[mi][ni], false, false);
        acc[mi][ni] = __builtin_amdgcn_wmma_f32_16x16x32_bf16(
            false, ah, false, bl[ni], (short)0, acc[mi][ni], false, false);
        acc[mi][ni] = __builtin_amdgcn_wmma_f32_16x16x32_bf16(
            false, al, false, bh[ni], (short)0, acc[mi][ni], false, false);
      }
    }
  }
}

// ---------------------------------------------------------------------------
// Kernel 1: H = x[b] @ W_in (256x64 tile, all timesteps of one batch in-block),
// leaky-integrator scan + tanh in LDS, write S as bf16 hi/lo planes.
// ---------------------------------------------------------------------------
__global__ __launch_bounds__(256) void k_fcin_scan(const __bf16* __restrict__ Xhi,
                                                   const __bf16* __restrict__ Xlo,
                                                   const __bf16* __restrict__ WtHi,
                                                   const __bf16* __restrict__ WtLo,
                                                   const float* __restrict__ b_in,
                                                   __bf16* __restrict__ Shi,
                                                   __bf16* __restrict__ Slo) {
  __shared__ __align__(16) float Hl[T_ * 64];  // 64 KB H tile [t][j]
  const int n0   = blockIdx.x * 64;
  const int b    = blockIdx.y;
  const int tid  = threadIdx.x;
  const int lane = tid & 31;
  const int wave = tid >> 5;
  const int wm = wave >> 1, wn = wave & 1;

  v8f acc[4][2];
  const v8f vzero = {0.f, 0.f, 0.f, 0.f, 0.f, 0.f, 0.f, 0.f};
#pragma unroll
  for (int mi = 0; mi < 4; ++mi)
#pragma unroll
    for (int ni = 0; ni < 2; ++ni) acc[mi][ni] = vzero;

  const size_t xoff = (size_t)b * T_ * DIN;
  gemm_tiles(Xhi + xoff, Xlo + xoff, DIN, DIN, WtHi, WtLo, n0, acc);

  // Scatter accumulators (H) to LDS: stride 64 => scan reads are conflict-free.
#pragma unroll
  for (int mi = 0; mi < 4; ++mi)
#pragma unroll
    for (int ni = 0; ni < 2; ++ni) {
      const int rbase = (wm * 4 + mi) * 16 + ((lane >> 4) << 3);
      const int c     = (wn * 2 + ni) * 16 + (lane & 15);
#pragma unroll
      for (int r = 0; r < 8; ++r) Hl[(rbase + r) * 64 + c] = acc[mi][ni][r];
    }
  __syncthreads();

  // Sequential membrane scan, one thread per column j.
  if (tid < 64) {
    const int j = tid;
    const float bias = b_in[n0 + j];
    float v = 0.f;
    const size_t base = (size_t)b * T_ * DHID + n0 + j;
#pragma unroll 1
    for (int t = 0; t < T_; ++t) {
      v = DECAYF * v + (Hl[t * 64 + j] + bias);
#if __has_builtin(__builtin_amdgcn_tanhf)
      const float s = __builtin_amdgcn_tanhf(v);
#else
      const float s = tanhf(v);
#endif
      __bf16 h, l;
      split1(s, h, l);
      Shi[base + (size_t)t * DHID] = h;
      Slo[base + (size_t)t * DHID] = l;
    }
  }
}

// ---------------------------------------------------------------------------
// Kernel 2: Y = S @ W_out + b_out. S hi/lo planes (128 MiB) are L2-resident.
// ---------------------------------------------------------------------------
__global__ __launch_bounds__(256) void k_fcout(const __bf16* __restrict__ Shi,
                                               const __bf16* __restrict__ Slo,
                                               const __bf16* __restrict__ WtHi,
                                               const __bf16* __restrict__ WtLo,
                                               const float* __restrict__ b_out,
                                               float* __restrict__ Y) {
  const int n0   = blockIdx.x * 64;
  const int m0   = blockIdx.y * 256;
  const int tid  = threadIdx.x;
  const int lane = tid & 31;
  const int wave = tid >> 5;
  const int wm = wave >> 1, wn = wave & 1;

  v8f acc[4][2];
  const v8f vzero = {0.f, 0.f, 0.f, 0.f, 0.f, 0.f, 0.f, 0.f};
#pragma unroll
  for (int mi = 0; mi < 4; ++mi)
#pragma unroll
    for (int ni = 0; ni < 2; ++ni) acc[mi][ni] = vzero;

  const size_t soff = (size_t)m0 * DHID;
  gemm_tiles(Shi + soff, Slo + soff, DHID, DHID, WtHi, WtLo, n0, acc);

#pragma unroll
  for (int mi = 0; mi < 4; ++mi)
#pragma unroll
    for (int ni = 0; ni < 2; ++ni) {
      const int colg = n0 + (wn * 2 + ni) * 16 + (lane & 15);
      const float bo = b_out[colg];
      const int rb = m0 + (wm * 4 + mi) * 16 + ((lane >> 4) << 3);
#pragma unroll
      for (int r = 0; r < 8; ++r)
        Y[(size_t)(rb + r) * DOUT + colg] = acc[mi][ni][r] + bo;
    }
}

// ---------------------------------------------------------------------------
// Probe kernels (compiled, never launched): exercise CDNA5 async paths so the
// toolchain validates the spellings. Used to guide next-round integration.
// ---------------------------------------------------------------------------
__global__ __launch_bounds__(64) void k_probe_async(const float* __restrict__ g,
                                                    float* __restrict__ o) {
  __shared__ __align__(16) float buf[256];
  unsigned lds = (unsigned)(uintptr_t)(&buf[(threadIdx.x & 15) * 4]);
  unsigned long long ga = (unsigned long long)(uintptr_t)(g + threadIdx.x * 4);
  asm volatile("global_load_async_to_lds_b128 %0, %1, off" ::"v"(lds), "v"(ga)
               : "memory");
#if __has_builtin(__builtin_amdgcn_s_wait_asynccnt)
  __builtin_amdgcn_s_wait_asynccnt(0);
#else
  asm volatile("s_wait_asynccnt 0" ::: "memory");
#endif
  __syncthreads();
  o[threadIdx.x] = buf[threadIdx.x];
}

#if __has_builtin(__builtin_amdgcn_tensor_load_to_lds)
typedef __attribute__((ext_vector_type(4))) unsigned int v4u;
typedef __attribute__((ext_vector_type(8))) int v8i;
typedef __attribute__((ext_vector_type(4))) int v4i;
__global__ __launch_bounds__(32) void k_probe_tdm() {
  // amdgpu-toolchain (clang-23) 6-arg form:
  // (uint32x4 g0, int32x8 g1, int32x4 g2, int32x4 g3, int32x8 g4, i32 cpol)
  v4u g0 = {0, 0, 0, 0};
  v8i g1 = {0, 0, 0, 0, 0, 0, 0, 0};
  v4i g2 = {0, 0, 0, 0};
  v4i g3 = {0, 0, 0, 0};
  v8i g4 = {0, 0, 0, 0, 0, 0, 0, 0};
  __builtin_amdgcn_tensor_load_to_lds(g0, g1, g2, g3, g4, 0);
#if __has_builtin(__builtin_amdgcn_s_wait_tensorcnt)
  __builtin_amdgcn_s_wait_tensorcnt(0);
#endif
}
#endif

// ---------------------------------------------------------------------------
extern "C" void kernel_launch(void* const* d_in, const int* in_sizes, int n_in,
                              void* d_out, int out_size, void* d_ws, size_t ws_size,
                              hipStream_t stream) {
  (void)in_sizes; (void)n_in; (void)out_size; (void)ws_size;
  const float* x     = (const float*)d_in[0];
  const float* W_in  = (const float*)d_in[1];
  const float* b_in  = (const float*)d_in[2];
  const float* W_out = (const float*)d_in[3];
  const float* b_out = (const float*)d_in[4];
  float* Y = (float*)d_out;

  // Workspace layout (total 208 MiB)
  char* w = (char*)d_ws;
  __bf16* Xhi = (__bf16*)w; w += (size_t)NTOK * DIN * 2;
  __bf16* Xlo = (__bf16*)w; w += (size_t)NTOK * DIN * 2;
  __bf16* WinHiT = (__bf16*)w; w += (size_t)DHID * DIN * 2;   // [N=2048][K=1024]
  __bf16* WinLoT = (__bf16*)w; w += (size_t)DHID * DIN * 2;
  __bf16* WoutHiT = (__bf16*)w; w += (size_t)DOUT * DHID * 2; // [N=1024][K=2048]
  __bf16* WoutLoT = (__bf16*)w; w += (size_t)DOUT * DHID * 2;
  __bf16* Shi = (__bf16*)w; w += (size_t)NTOK * DHID * 2;
  __bf16* Slo = (__bf16*)w;

  // One-time operand preparation
  const int n8 = NTOK * DIN / 8;
  k_split_x<<<dim3(n8 / 256), 256, 0, stream>>>(x, Xhi, Xlo, n8);
  k_split_wt<<<dim3(DHID / 64, DIN / 64), 256, 0, stream>>>(W_in, DIN, DHID, WinHiT, WinLoT);
  k_split_wt<<<dim3(DOUT / 64, DHID / 64), 256, 0, stream>>>(W_out, DHID, DOUT, WoutHiT, WoutLoT);

  // Fused fc_in + scan + tanh, then fc_out
  k_fcin_scan<<<dim3(DHID / 64, B_), 256, 0, stream>>>(Xhi, Xlo, WinHiT, WinLoT, b_in, Shi, Slo);
  k_fcout<<<dim3(DOUT / 64, NTOK / 256), 256, 0, stream>>>(Shi, Slo, WoutHiT, WoutLoT, b_out, Y);
}
